// StructuredConnection_52209622450452
// MI455X (gfx1250) — compile-verified, matrix operand order checked by reading, not run on hardware
//
#include <hip/hip_runtime.h>

// ---------------------------------------------------------------------------
// StructuredConnection (banded 98%-sparse linear) for gfx1250 / MI455X.
// out[b,o] = sum_{j<81} x[b,(81o+j)%4096] * w[o,(81o+j)%4096] + bias[o]
//
// Output permutation trick: s(o)=81*o mod 4096 is a bijection (inv = 2225).
// Outputs with band-starts s in [16g,16g+16) all read x columns in the
// contiguous window [16g, 16g+96) -> 16x16x96 WMMA tiles with only ~1.19x
// padding. Kernel is HBM-bound (~536MB @ 23.3 TB/s ~= 23us).
// ---------------------------------------------------------------------------

typedef __attribute__((ext_vector_type(16))) _Float16 v16h;
typedef __attribute__((ext_vector_type(8)))  _Float16 v8h;
typedef __attribute__((ext_vector_type(4)))  _Float16 v4h;
typedef __attribute__((ext_vector_type(8)))  float    v8f;

#define IN_F     4096
#define OUT_F    4096
#define N_BATCH  16384
#define INV81    2225        // 81^-1 mod 4096
#define NGROUPS  256         // 4096 / 16 outputs per group
#define KSPAN    96          // 16 + 80 band span, = 3 * 32 (f16 WMMA K)
#define KSTEPS   3
#define BROWS    128         // batch rows per block (8 waves * 16)
#define ROWPAD   104         // f16 per LDS row (13*16B stride, conflict-free)

// 16-bit fragment K index for (lane, vgpr r, half h) per CDNA5 ISA 7.12.2:
// lanes 0-15: K = {0..7} in v0-3, {16..23} in v4-7; lanes 16-31: +8 / +24.
__device__ __forceinline__ int frag_k(int lane, int r, int h) {
    return ((lane >> 4) * 8) + ((r & 3) * 2) + ((r >> 2) * 16) + h;
}

// ---------------------------------------------------------------------------
// Kernel 1: pack banded masked weights into per-(group, kstep, lane) WMMA
// B-fragments (f16). Layout: bpack[((g*3 + t)*32 + lane) * 8 dwords].
// ---------------------------------------------------------------------------
__global__ void __launch_bounds__(256) pack_b_kernel(
    const float* __restrict__ w, unsigned int* __restrict__ bpack) {
    int tid  = blockIdx.x * 256 + threadIdx.x;   // (g*3 + t)*32 + lane
    int lane = tid & 31;
    int gt   = tid >> 5;
    int t    = gt % KSTEPS;
    int g    = gt / KSTEPS;
    if (g >= NGROUPS) return;

    int n = lane & 15;                                 // B column (output slot)
    int o = (INV81 * (g * 16 + n)) & (OUT_F - 1);      // real output index
    unsigned int pk[8];
#pragma unroll
    for (int r = 0; r < 8; ++r) {
        unsigned int d = 0;
#pragma unroll
        for (int h = 0; h < 2; ++h) {
            int k = t * 32 + frag_k(lane, r, h);       // k within 96-window
            float v = 0.0f;
            if (k >= n && k <= n + 80) {               // inside o's 81-band
                int i = (g * 16 + k) & (IN_F - 1);     // wrapped input col
                v = w[(size_t)o * IN_F + i];
            }
            _Float16 hv = (_Float16)v;
            d |= ((unsigned int)__builtin_bit_cast(unsigned short, hv)) << (16 * h);
        }
        pk[r] = d;
    }
    uint4* dst = (uint4*)(bpack + (size_t)tid * 8);
    dst[0] = make_uint4(pk[0], pk[1], pk[2], pk[3]);
    dst[1] = make_uint4(pk[4], pk[5], pk[6], pk[7]);
}

// ---------------------------------------------------------------------------
// Kernel 2: banded WMMA GEMM. Block = 256 threads (8 waves), handles one
// group g (16 outputs) x 128 batch rows. K = 96 (3 wmma steps).
// ---------------------------------------------------------------------------
__global__ void __launch_bounds__(256) band_wmma_kernel(
    const float* __restrict__ x, const unsigned int* __restrict__ bpack,
    const float* __restrict__ bias, float* __restrict__ out) {
    const int g  = blockIdx.x;
    const int b0 = blockIdx.y * BROWS;

    __shared__ _Float16 sx[BROWS * ROWPAD];   // 26.6 KB, f16 x-tile

    // ---- Stage x[b0..b0+127][16g .. 16g+96) -> LDS as f16 (float4 granules)
    const int colbase = g * 16;
#pragma unroll
    for (int e = threadIdx.x; e < BROWS * (KSPAN / 4); e += 256) {
        int row = e / (KSPAN / 4);
        int k4  = e % (KSPAN / 4);
        int i   = (colbase + k4 * 4) & (IN_F - 1);     // wrap: i%4==0 -> safe
        float4 v = *(const float4*)(x + (size_t)(b0 + row) * IN_F + i);
        v4h hv = { (_Float16)v.x, (_Float16)v.y, (_Float16)v.z, (_Float16)v.w };
        *(v4h*)(&sx[row * ROWPAD + k4 * 4]) = hv;      // ds_store_b64
    }
    __syncthreads();

    const int lane  = threadIdx.x & 31;
    const int wave  = threadIdx.x >> 5;                // 0..7 -> 16 rows each
    const int mrow  = lane & 15;
    const int khalf = lane >> 4;                       // K lane-half select

    const _Float16* arow = &sx[(wave * 16 + mrow) * ROWPAD];
    const v16h* bfrag = (const v16h*)bpack;            // 32B-aligned frags

    v8f acc = {};
#pragma unroll
    for (int t = 0; t < KSTEPS; ++t) {
        // A fragment: two ds_load_b128 (K 0..7 / 16..23 for lanes 0-15, +8 hi)
        v8h alo = *(const v8h*)(arow + t * 32 + khalf * 8);
        v8h ahi = *(const v8h*)(arow + t * 32 + khalf * 8 + 16);
        v16h a  = __builtin_shufflevector(alo, ahi,
                    0,1,2,3,4,5,6,7,8,9,10,11,12,13,14,15);
        // B fragment: 32B from packed weights (L2-resident, shared by grid.y)
        v16h b  = bfrag[(size_t)(g * KSTEPS + t) * 32 + lane];
        acc = __builtin_amdgcn_wmma_f32_16x16x32_f16(
                  false, a, false, b, (short)0, acc, false, false);
    }

    // ---- Bias + scattered store (o permuted; L2 combines the 4B stores)
    const int   n  = lane & 15;
    const int   o  = (INV81 * (g * 16 + n)) & (OUT_F - 1);
    const float bz = bias[o];
    float* orow = out + (size_t)(b0 + wave * 16 + khalf * 8) * OUT_F + o;
#pragma unroll
    for (int r = 0; r < 8; ++r)                        // VGPR r -> M = r(+8)
        orow[(size_t)r * OUT_F] = acc[r] + bz;
}

// ---------------------------------------------------------------------------
extern "C" void kernel_launch(void* const* d_in, const int* in_sizes, int n_in,
                              void* d_out, int out_size, void* d_ws, size_t ws_size,
                              hipStream_t stream) {
    const float* x      = (const float*)d_in[0];   // [16384, 4096]
    const float* weight = (const float*)d_in[1];   // [4096, 4096]
    const float* bias   = (const float*)d_in[2];   // [4096]
    // d_in[3] = mask, reproduced analytically (banded pattern)
    float* out = (float*)d_out;
    unsigned int* bpack = (unsigned int*)d_ws;     // 256*3*32*8*4 = 768 KB

    // Kernel 1: pack weight band -> WMMA B fragments (24576 lanes)
    pack_b_kernel<<<dim3((NGROUPS * KSTEPS * 32 + 255) / 256), 256, 0, stream>>>(
        weight, bpack);

    // Kernel 2: grid.x = group (fast -> L2 sharing of x windows + out lines),
    //           grid.y = batch tile of 128 rows
    band_wmma_kernel<<<dim3(NGROUPS, N_BATCH / BROWS), 256, 0, stream>>>(
        x, bpack, bias, out);
}
